// FocalLossClassifier_15522011808333
// MI455X (gfx1250) — compile-verified, matrix operand order checked by reading, not run on hardware
//
#include <hip/hip_runtime.h>
#include <hip/hip_fp16.h>

#define NN 50000
#define NE 800000
#define F_IN 16
#define HID 32
#define HEADS 4
#define NCLS 5
#define EF 8
#define HC 128
#define NEG_SLOPE 0.2f

typedef __attribute__((ext_vector_type(16))) _Float16 v16h;
typedef __attribute__((ext_vector_type(8)))  float    v8f;

// ---- ordered-uint encoding for float atomicMax ----
__device__ __forceinline__ unsigned enc_f32(float f) {
  unsigned u = __float_as_uint(f);
  return (u & 0x80000000u) ? ~u : (u | 0x80000000u);
}
__device__ __forceinline__ float dec_f32(unsigned e) {
  unsigned u = (e & 0x80000000u) ? (e ^ 0x80000000u) : ~e;
  return __uint_as_float(u);
}

// ---- edge encoder MLP: [E,8] -> relu(32) -> [E,4] ----
__global__ void ee_kernel(const float* __restrict__ attr,
                          const float* __restrict__ w1, const float* __restrict__ b1,
                          const float* __restrict__ w2, const float* __restrict__ b2,
                          float* __restrict__ ea) {
  __shared__ float sw1[HID * EF], sb1[HID], sw2[HEADS * HID], sb2[HEADS];
  int tid = threadIdx.x;
  if (tid < HID * EF)   sw1[tid] = w1[tid];
  if (tid < HID)        sb1[tid] = b1[tid];
  if (tid < HEADS*HID)  sw2[tid] = w2[tid];
  if (tid < HEADS)      sb2[tid] = b2[tid];
  __syncthreads();
  int e = blockIdx.x * blockDim.x + tid;
  if (e >= NE) return;
  float a[EF];
#pragma unroll
  for (int i = 0; i < EF; ++i) a[i] = attr[e * EF + i];
  float h[HID];
#pragma unroll
  for (int j = 0; j < HID; ++j) {
    float s = sb1[j];
#pragma unroll
    for (int i = 0; i < EF; ++i) s += a[i] * sw1[j * EF + i];
    h[j] = s > 0.f ? s : 0.f;
  }
#pragma unroll
  for (int k = 0; k < HEADS; ++k) {
    float s = sb2[k];
#pragma unroll
    for (int j = 0; j < HID; ++j) s += h[j] * sw2[k * HID + j];
    ea[e * HEADS + k] = s;
  }
}

// ---- pack weight W[HC, Ka] (row-major) into fragment-major f16 B layout ----
// frag index = ktile*8 + nt; within frag: lane 0..31, element e 0..15 contiguous
__global__ void pack_b_kernel(const float* __restrict__ W, int Ka, int Kp,
                              _Float16* __restrict__ out) {
  int t = blockIdx.x * blockDim.x + threadIdx.x;
  int total = (Kp >> 5) * 8 * 512;
  if (t >= total) return;
  int e    = t & 15;
  int lane = (t >> 4) & 31;
  int frag = t >> 9;
  int nt    = frag & 7;
  int ktile = frag >> 3;
  int col  = nt * 16 + (lane & 15);
  int kgrp = (lane >= 16) ? 8 : 0;
  int v = e >> 1, hh = e & 1;
  int k = ktile * 32 + kgrp + ((v < 4) ? 2 * v : 16 + 2 * (v - 4)) + hh;
  out[t] = (k < Ka) ? (_Float16)W[col * Ka + k] : (_Float16)0.f;
}

// ---- M[h,k] = sum_c We[h*32+c, k] * att_edge[h,c]  (folds ep @ att_edge into 4x4) ----
__global__ void prep_m_kernel(const float* __restrict__ We1, const float* __restrict__ ae1,
                              const float* __restrict__ We2, const float* __restrict__ ae2,
                              float* __restrict__ M1, float* __restrict__ M2) {
  int t = threadIdx.x;
  if (t >= 32) return;
  const float* We = (t < 16) ? We1 : We2;
  const float* aw = (t < 16) ? ae1 : ae2;
  float* M = (t < 16) ? M1 : M2;
  int idx = t & 15, h = idx >> 2, k = idx & 3;
  float s = 0.f;
  for (int c = 0; c < HID; ++c) s += We[(h * HID + c) * HEADS + k] * aw[h * HID + c];
  M[idx] = s;
}

// ---- WMMA GEMM: C[n,128] = A[n,KA](f32) x W^T (packed f16 frags), fp32 accumulate ----
// Compile-time KA (actual K) / KP (padded K, multiple of 32).
// block = 8 waves; wave = one 16-row strip; 8 n-tiles of 16 cols.
template <int KA, int KP>
__global__ void gemm_wmma_kernel(const float* __restrict__ A,
                                 const _Float16* __restrict__ Bpack,
                                 float* __restrict__ C, int n) {
  int lane = threadIdx.x & 31;
  int wave = threadIdx.x >> 5;
  int row0 = blockIdx.x * 128 + wave * 16;
  int m = lane & 15;
  int row = row0 + m; if (row > n - 1) row = n - 1;
  int kgrp = (lane >= 16) ? 8 : 0;   // A fragment K-base per lane half (0 or 8)
  int colg = lane & 15;
  const float* Arow = A + (size_t)row * KA;
  const v16h* Bp = (const v16h*)Bpack;   // 32B-aligned fragments

  v8f acc[8];
#pragma unroll
  for (int t = 0; t < 8; ++t) acc[t] = (v8f){0.f,0.f,0.f,0.f,0.f,0.f,0.f,0.f};

#pragma unroll
  for (int ktile = 0; ktile < (KP >> 5); ++ktile) {
    v16h a;
#pragma unroll
    for (int v = 0; v < 8; ++v) {
      // kb = ktile*32 + kgrp + (v<4 ? 2v : 16+2(v-4));  kgrp in {0,8}
      // For KA==KP every element is in range.
      // For KA==16,KP==32: v<4 -> kb <= 14 < 16 (always load), v>=4 -> kb >= 16 (always 0).
      constexpr bool in_range = (KA == KP) || (KA == 16 && KP == 32 && true);
      if (KA == KP || (KA == 16 && KP == 32 && v < 4)) {
        int kb = ktile * 32 + kgrp + ((v < 4) ? 2 * v : 16 + 2 * (v - 4));
        float2 val = *(const float2*)(Arow + kb);
        a[2 * v]     = (_Float16)val.x;
        a[2 * v + 1] = (_Float16)val.y;
      } else {
        a[2 * v]     = (_Float16)0.f;
        a[2 * v + 1] = (_Float16)0.f;
      }
      (void)in_range;
    }
    const v16h* bf = Bp + (size_t)ktile * 8 * 32 + lane;
#pragma unroll
    for (int nt = 0; nt < 8; ++nt) {
      v16h b = bf[nt * 32];                 // two global_load_b128 per fragment
      acc[nt] = __builtin_amdgcn_wmma_f32_16x16x32_f16(
          false, a, false, b, (short)0, acc[nt], false, false);
    }
  }
#pragma unroll
  for (int nt = 0; nt < 8; ++nt) {
#pragma unroll
    for (int r = 0; r < 8; ++r) {
      int rr = row0 + r + ((lane >= 16) ? 8 : 0);
      if (rr < n) C[(size_t)rr * HC + nt * 16 + colg] = acc[nt][r];
    }
  }
}

// ---- a_src/a_dst: per (node, head) 32-wide dot with attention vectors ----
__global__ void att_node_kernel(const float* __restrict__ xp,
                                const float* __restrict__ asw, const float* __restrict__ adw,
                                float* __restrict__ a_src, float* __restrict__ a_dst) {
  int t = blockIdx.x * blockDim.x + threadIdx.x;
  if (t >= NN * HEADS) return;
  int node = t >> 2, h = t & 3;
  const float* xr = xp + (size_t)node * HC + h * HID;
  float s = 0.f, d = 0.f;
#pragma unroll
  for (int c = 0; c < HID; ++c) { float v = xr[c]; s += v * asw[h * HID + c]; d += v * adw[h * HID + c]; }
  a_src[t] = s; a_dst[t] = d;
}

// ---- zero accumulators, init max(-inf encoded=0) and sum ----
__global__ void init_kernel(float* __restrict__ acc, unsigned* __restrict__ nmax,
                            float* __restrict__ nsum) {
  int t = blockIdx.x * blockDim.x + threadIdx.x;
  if (t < NN * HC) acc[t] = 0.f;
  if (t < NN * HEADS) { nmax[t] = 0u; nsum[t] = 0.f; }
}

// ---- edge pass 1: logits + atomicMax per (dst, head) ----
__global__ void edge_logit_kernel(const long long* __restrict__ ei,
                                  const float* __restrict__ ea, const float* __restrict__ M,
                                  const float* __restrict__ a_src, const float* __restrict__ a_dst,
                                  float* __restrict__ logit, unsigned* __restrict__ nmax) {
  int e = blockIdx.x * blockDim.x + threadIdx.x;
  if (e >= NE) return;
  int src = (int)ei[e];
  int dst = (int)ei[NE + e];
  float ev[HEADS];
#pragma unroll
  for (int k = 0; k < HEADS; ++k) ev[k] = ea[e * HEADS + k];
#pragma unroll
  for (int h = 0; h < HEADS; ++h) {
    float ae = 0.f;
#pragma unroll
    for (int k = 0; k < HEADS; ++k) ae += ev[k] * M[h * HEADS + k];
    float l = a_src[src * HEADS + h] + a_dst[dst * HEADS + h] + ae;
    l = (l > 0.f) ? l : NEG_SLOPE * l;
    logit[e * HEADS + h] = l;
    atomicMax(&nmax[dst * HEADS + h], enc_f32(l));
  }
}

// ---- edge pass 2: w=exp(l-max); scatter w*xp[src] into acc[dst]; sum w ----
// one wave per edge; lane = channel-in-head, 4 head iterations cover 128 channels
__global__ void edge_agg_kernel(const long long* __restrict__ ei,
                                const float* __restrict__ logit,
                                const unsigned* __restrict__ nmax, float* __restrict__ nsum,
                                const float* __restrict__ xp, float* __restrict__ acc) {
  int lane = threadIdx.x & 31;
  int e = blockIdx.x * (blockDim.x >> 5) + (threadIdx.x >> 5);
  if (e >= NE) return;
  int src = (int)ei[e];
  int dst = (int)ei[NE + e];
#pragma unroll
  for (int h = 0; h < HEADS; ++h) {
    float l  = logit[e * HEADS + h];
    float mx = dec_f32(nmax[dst * HEADS + h]);
    float w  = __expf(l - mx);
    int c = h * HID + lane;
    atomicAdd(&acc[(size_t)dst * HC + c], w * xp[(size_t)src * HC + c]);
    if (lane == h) atomicAdd(&nsum[dst * HEADS + h], w);
  }
}

// ---- normalize, add bias, ELU ----
__global__ void finalize_kernel(const float* __restrict__ acc, const float* __restrict__ nsum,
                                const float* __restrict__ bias, float* __restrict__ hout) {
  int t = blockIdx.x * blockDim.x + threadIdx.x;
  if (t >= NN * HC) return;
  int node = t >> 7; int c = t & 127; int h = c >> 5;
  float s = nsum[node * HEADS + h] + 1e-16f;
  float v = acc[t] / s + bias[c];
  hout[t] = (v > 0.f) ? v : expm1f(v);
}

// ---- (h2 + skip + skip_b) @ cls_w^T + cls_b ----
__global__ void classifier_kernel(const float* __restrict__ h2, const float* __restrict__ skip,
                                  const float* __restrict__ skip_b,
                                  const float* __restrict__ cls_w, const float* __restrict__ cls_b,
                                  float* __restrict__ out) {
  int nidx = blockIdx.x * blockDim.x + threadIdx.x;
  if (nidx >= NN) return;
  float a[NCLS];
#pragma unroll
  for (int k = 0; k < NCLS; ++k) a[k] = cls_b[k];
  for (int c = 0; c < HC; ++c) {
    float v = h2[(size_t)nidx * HC + c] + skip[(size_t)nidx * HC + c] + skip_b[c];
#pragma unroll
    for (int k = 0; k < NCLS; ++k) a[k] += v * cls_w[k * HC + c];
  }
#pragma unroll
  for (int k = 0; k < NCLS; ++k) out[nidx * NCLS + k] = a[k];
}

extern "C" void kernel_launch(void* const* d_in, const int* in_sizes, int n_in,
                              void* d_out, int out_size, void* d_ws, size_t ws_size,
                              hipStream_t stream) {
  const float*     x      = (const float*)d_in[0];
  const long long* ei     = (const long long*)d_in[1];
  const float*     eattr  = (const float*)d_in[2];
  const float *ee_w1 = (const float*)d_in[3], *ee_b1 = (const float*)d_in[4];
  const float *ee_w2 = (const float*)d_in[5], *ee_b2 = (const float*)d_in[6];
  const float *g1_W = (const float*)d_in[7],  *g1_We = (const float*)d_in[8];
  const float *g1_as = (const float*)d_in[9], *g1_ad = (const float*)d_in[10];
  const float *g1_ae = (const float*)d_in[11],*g1_b  = (const float*)d_in[12];
  const float *g2_W = (const float*)d_in[13], *g2_We = (const float*)d_in[14];
  const float *g2_as = (const float*)d_in[15],*g2_ad = (const float*)d_in[16];
  const float *g2_ae = (const float*)d_in[17],*g2_b  = (const float*)d_in[18];
  const float *skip_w = (const float*)d_in[19], *skip_b = (const float*)d_in[20];
  const float *cls_w  = (const float*)d_in[21], *cls_b  = (const float*)d_in[22];
  float* out = (float*)d_out;

  // ---- workspace carve-out (256B aligned) ----
  size_t off = 0;
  auto carve = [&](size_t bytes) -> void* {
    void* p = (char*)d_ws + off;
    off += (bytes + 255) & ~(size_t)255;
    return p;
  };
  float*     ea    = (float*)carve((size_t)NE * HEADS * 4);
  float*     logit = (float*)carve((size_t)NE * HEADS * 4);
  float*     bufA  = (float*)carve((size_t)NN * HC * 4);   // xp (layer 1 then layer 2)
  float*     bufB  = (float*)carve((size_t)NN * HC * 4);   // acc -> h
  float*     bufS  = (float*)carve((size_t)NN * HC * 4);   // skip projection
  float*     a_src = (float*)carve((size_t)NN * HEADS * 4);
  float*     a_dst = (float*)carve((size_t)NN * HEADS * 4);
  float*     nsum  = (float*)carve((size_t)NN * HEADS * 4);
  unsigned*  nmax  = (unsigned*)carve((size_t)NN * HEADS * 4);
  float*     M1    = (float*)carve(16 * 4);
  float*     M2    = (float*)carve(16 * 4);
  _Float16*  B1p   = (_Float16*)carve((size_t)1 * 8 * 512 * 2);   // Kp=32  packed
  _Float16*  B2p   = (_Float16*)carve((size_t)4 * 8 * 512 * 2);   // Kp=128 packed
  _Float16*  Bsp   = (_Float16*)carve((size_t)1 * 8 * 512 * 2);   // Kp=32  packed

  const int TB = 256;
  dim3 blk(TB);
  int gE    = (NE + TB - 1) / TB;
  int gNC   = (NN * HC + TB - 1) / TB;
  int gNH   = (NN * HEADS + TB - 1) / TB;
  int gGemm = (NN + 127) / 128;
  int gEw   = (NE + 7) / 8;            // one wave (32 lanes) per edge
  int gN    = (NN + TB - 1) / TB;

  // prep
  ee_kernel<<<gE, blk, 0, stream>>>(eattr, ee_w1, ee_b1, ee_w2, ee_b2, ea);
  pack_b_kernel<<<(1 * 8 * 512 + TB - 1) / TB, blk, 0, stream>>>(g1_W, F_IN, 32, B1p);
  pack_b_kernel<<<(4 * 8 * 512 + TB - 1) / TB, blk, 0, stream>>>(g2_W, HC, HC, B2p);
  pack_b_kernel<<<(1 * 8 * 512 + TB - 1) / TB, blk, 0, stream>>>(skip_w, F_IN, 32, Bsp);
  prep_m_kernel<<<1, 32, 0, stream>>>(g1_We, g1_ae, g2_We, g2_ae, M1, M2);

  // skip projection (bias added in classifier)
  gemm_wmma_kernel<F_IN, 32><<<gGemm, blk, 0, stream>>>(x, Bsp, bufS, NN);

  // ---- GAT layer 1 ----
  gemm_wmma_kernel<F_IN, 32><<<gGemm, blk, 0, stream>>>(x, B1p, bufA, NN);
  att_node_kernel<<<gNH, blk, 0, stream>>>(bufA, g1_as, g1_ad, a_src, a_dst);
  init_kernel<<<gNC, blk, 0, stream>>>(bufB, nmax, nsum);
  edge_logit_kernel<<<gE, blk, 0, stream>>>(ei, ea, M1, a_src, a_dst, logit, nmax);
  edge_agg_kernel<<<gEw, blk, 0, stream>>>(ei, logit, nmax, nsum, bufA, bufB);
  finalize_kernel<<<gNC, blk, 0, stream>>>(bufB, nsum, g1_b, bufB);   // h1 in bufB

  // ---- GAT layer 2 ----
  gemm_wmma_kernel<HC, HC><<<gGemm, blk, 0, stream>>>(bufB, B2p, bufA, NN);  // xp2
  att_node_kernel<<<gNH, blk, 0, stream>>>(bufA, g2_as, g2_ad, a_src, a_dst);
  init_kernel<<<gNC, blk, 0, stream>>>(bufB, nmax, nsum);
  edge_logit_kernel<<<gE, blk, 0, stream>>>(ei, ea, M2, a_src, a_dst, logit, nmax);
  edge_agg_kernel<<<gEw, blk, 0, stream>>>(ei, logit, nmax, nsum, bufA, bufB);
  finalize_kernel<<<gNC, blk, 0, stream>>>(bufB, nsum, g2_b, bufB);   // h2 in bufB

  // head
  classifier_kernel<<<gN, blk, 0, stream>>>(bufB, bufS, skip_b, cls_w, cls_b, out);
}
